// TransfomerAttentionLayer_6064493822585
// MI455X (gfx1250) — compile-verified, matrix operand order checked by reading, not run on hardware
//
#include <hip/hip_runtime.h>
#include <hip/hip_bf16.h>

typedef __attribute__((ext_vector_type(16))) __bf16 v16bf;
typedef __attribute__((ext_vector_type(8)))  float  v8f;

// ---------- helpers ----------
__device__ __forceinline__ unsigned fkey(float f) {
    unsigned u = __float_as_uint(f);
    return (u & 0x80000000u) ? ~u : (u | 0x80000000u);   // order-preserving map
}
__device__ __forceinline__ float funkey(unsigned k) {
    if (k == 0u) return 0.0f;                            // untouched segment -> 0 (matches ref isfinite clamp)
    unsigned u = (k & 0x80000000u) ? (k ^ 0x80000000u) : ~k;
    return __uint_as_float(u);
}
// ISA 16-bit A/B fragment K mapping: lanes 0-15 hold K {g*8+e, 16+g*8+e}, g = lane>>4
__device__ __forceinline__ int fragK(int e, int lane) {
    int g = lane >> 4;
    return (e < 8) ? (g * 8 + e) : (16 + g * 8 + (e - 8));
}

// ---------- prep: swizzle fp32 weight (row-major [128 out][srcStride in]) into bf16 WMMA-B fragments ----------
__global__ __launch_bounds__(256) void swizzle_kernel(const float* __restrict__ W, int srcStride, int kOff,
                                                      __bf16* __restrict__ dst) {
    int f = blockIdx.x;                 // f = chunk*8 + nTile
    int chunk = f >> 3, nTile = f & 7;
    for (int idx = threadIdx.x; idx < 512; idx += 256) {
        int lane = idx >> 4, e = idx & 15;
        int k = chunk * 32 + fragK(e, lane);
        int n = nTile * 16 + (lane & 15);
        dst[(size_t)f * 512 + lane * 16 + e] = (__bf16)W[(size_t)n * srcStride + kOff + k];
    }
}

__global__ void qconst_kernel(const float* __restrict__ Wq, const float* __restrict__ bq,
                              const float* __restrict__ time_b, float* __restrict__ qconst) {
    int o = threadIdx.x;
    if (o < 128) {
        float s = bq[o];
        for (int j = 0; j < 128; ++j) s += Wq[(size_t)o * 256 + 128 + j] * __cosf(time_b[j]);
        qconst[o] = s;
    }
}

__global__ void zero_kernel(float* __restrict__ p, int n) {
    int i = blockIdx.x * 256 + threadIdx.x;
    if (i < n) p[i] = 0.0f;
}

// ---------- Q projection: Qn = tgt @ Wq[:, :128].T + qconst ----------
__global__ __launch_bounds__(256) void q_kernel(const float* __restrict__ node_h, const __bf16* __restrict__ wq_sw,
                                                const float* __restrict__ qconst, float* __restrict__ Qn, int nDst) {
    __shared__ __bf16 aT[16 * 128];
    int tid = threadIdx.x;
    int r0 = blockIdx.x * 16;
    for (int idx = tid; idx < 16 * 128; idx += 256) {
        int row = idx >> 7, col = idx & 127;
        int rr = r0 + row; if (rr >= nDst) rr = nDst - 1;
        aT[idx] = (__bf16)node_h[(size_t)rr * 128 + col];
    }
    __syncthreads();
    int wave = tid >> 5, lane = tid & 31, g = lane >> 4, rowm = lane & 15;
    v8f c = {};
    for (int kc = 0; kc < 4; ++kc) {
        v16bf a, b;
        int kb = kc * 32;
        #pragma unroll
        for (int e = 0; e < 8; ++e) {
            a[e]     = aT[rowm * 128 + kb + g * 8 + e];
            a[e + 8] = aT[rowm * 128 + kb + 16 + g * 8 + e];
        }
        const __bf16* pb = wq_sw + ((size_t)(kc * 8 + wave) * 32 + lane) * 16;
        #pragma unroll
        for (int e = 0; e < 16; ++e) b[e] = pb[e];
        c = __builtin_amdgcn_wmma_f32_16x16x32_bf16(false, a, false, b, (short)0, c, false, false);
    }
    int ncol = wave * 16 + (lane & 15);
    float qc = qconst[ncol];
    #pragma unroll
    for (int r = 0; r < 8; ++r) {
        int row = r0 + r + g * 8;
        if (row < nDst) Qn[(size_t)row * 128 + ncol] = c[r] + qc;
    }
}

// ---------- fused: time-encode + K/V GEMM + attention score + segment max ----------
__global__ __launch_bounds__(256) void kv_score_kernel(
    const float* __restrict__ node_h, const float* __restrict__ edge_f, const float* __restrict__ dt,
    const int* __restrict__ dst_idx, const float* __restrict__ time_w, const float* __restrict__ time_b,
    const float* __restrict__ bk, const float* __restrict__ bv,
    const __bf16* __restrict__ wk_sw, const __bf16* __restrict__ wv_sw,
    const float* __restrict__ Qn, float* __restrict__ score, unsigned* __restrict__ mKey,
    __bf16* __restrict__ Vbf, int nDst, int nEdges) {
    __shared__ __bf16 kvT[16 * 384];
    __shared__ float  qT[16 * 128];
    __shared__ int    dstT[16];
    __shared__ float  scT[32];
    int tid = threadIdx.x;
    int e0 = blockIdx.x * 16;
    if (tid < 16) {
        int e = e0 + tid; if (e >= nEdges) e = nEdges - 1;
        dstT[tid] = dst_idx[e];
    }
    if (tid < 32) scT[tid] = 0.0f;
    __syncthreads();
    for (int idx = tid; idx < 16 * 384; idx += 256) {
        int row = idx / 384, col = idx - row * 384;
        int e = e0 + row; if (e >= nEdges) e = nEdges - 1;
        float v;
        if (col < 128)      v = node_h[(size_t)(nDst + e) * 128 + col];
        else if (col < 256) v = edge_f[(size_t)e * 128 + (col - 128)];
        else { int j = col - 256; v = __cosf(dt[e] * time_w[j] + time_b[j]); }
        kvT[idx] = (__bf16)v;
    }
    for (int idx = tid; idx < 16 * 128; idx += 256) {
        int row = idx >> 7, col = idx & 127;
        qT[idx] = Qn[(size_t)dstT[row] * 128 + col];
    }
    __syncthreads();

    int wave = tid >> 5, lane = tid & 31, g = lane >> 4, rowm = lane & 15;
    int nTile = wave;
    v8f ck = {}, cv = {};
    for (int kc = 0; kc < 12; ++kc) {
        v16bf a, bK, bV;
        int kb = kc * 32;
        #pragma unroll
        for (int e = 0; e < 8; ++e) {
            a[e]     = kvT[rowm * 384 + kb + g * 8 + e];
            a[e + 8] = kvT[rowm * 384 + kb + 16 + g * 8 + e];
        }
        const __bf16* pk = wk_sw + ((size_t)(kc * 8 + nTile) * 32 + lane) * 16;
        const __bf16* pv = wv_sw + ((size_t)(kc * 8 + nTile) * 32 + lane) * 16;
        #pragma unroll
        for (int e = 0; e < 16; ++e) { bK[e] = pk[e]; bV[e] = pv[e]; }
        ck = __builtin_amdgcn_wmma_f32_16x16x32_bf16(false, a, false, bK, (short)0, ck, false, false);
        cv = __builtin_amdgcn_wmma_f32_16x16x32_bf16(false, a, false, bV, (short)0, cv, false, false);
    }
    int ncol = nTile * 16 + (lane & 15);
    float bkv = bk[ncol], bvv = bv[ncol];
    int head = nTile >> 2;
    float part[8];
    #pragma unroll
    for (int r = 0; r < 8; ++r) {
        int M = r + g * 8;                               // C layout: lanes 0-15 -> M=r, lanes 16-31 -> M=8+r
        float kval = ck[r] + bkv;
        part[r] = kval * qT[M * 128 + ncol];
        float vval = cv[r] + bvv;
        if (e0 + M < nEdges) Vbf[(size_t)(e0 + M) * 128 + ncol] = (__bf16)vval;
    }
    #pragma unroll
    for (int r = 0; r < 8; ++r) {
        float v = part[r];
        v += __shfl_xor(v, 1);
        v += __shfl_xor(v, 2);
        v += __shfl_xor(v, 4);
        v += __shfl_xor(v, 8);
        if ((lane & 15) == 0) atomicAdd(&scT[(r + g * 8) * 2 + head], v);
    }
    __syncthreads();
    if (tid < 32) {
        int m = tid >> 1, h = tid & 1;
        if (e0 + m < nEdges) {
            float s = scT[m * 2 + h];
            s = (s >= 0.0f) ? s : 0.2f * s;              // leaky relu
            score[(size_t)(e0 + m) * 2 + h] = s;
            atomicMax(&mKey[(size_t)dstT[m] * 2 + h], fkey(s));
        }
    }
}

// ---------- segment softmax: e = exp(score - m), s = segment_sum(e) ----------
__global__ __launch_bounds__(256) void exp_kernel(const float* __restrict__ score, const int* __restrict__ dst_idx,
                                                  const unsigned* __restrict__ mKey, float* __restrict__ eArr,
                                                  float* __restrict__ sSum, int nEdges) {
    int t = blockIdx.x * 256 + threadIdx.x;
    if (t < nEdges * 2) {
        int edge = t >> 1, h = t & 1;
        int d = dst_idx[edge];
        float m = funkey(mKey[(size_t)d * 2 + h]);
        float ev = __expf(score[t] - m);
        eArr[t] = ev;
        atomicAdd(&sSum[(size_t)d * 2 + h], ev);
    }
}

// ---------- weighted scatter: h[dst] += att * V ----------
__global__ __launch_bounds__(256) void agg_kernel(const float* __restrict__ eArr, const float* __restrict__ sSum,
                                                  const int* __restrict__ dst_idx, const __bf16* __restrict__ Vbf,
                                                  float* __restrict__ hAgg, int nEdges) {
    int wave = threadIdx.x >> 5, lane = threadIdx.x & 31;
    int edge = blockIdx.x * 8 + wave;
    if (edge >= nEdges) return;
    int d = dst_idx[edge];
    float a0 = eArr[(size_t)edge * 2 + 0] / sSum[(size_t)d * 2 + 0];
    float a1 = eArr[(size_t)edge * 2 + 1] / sSum[(size_t)d * 2 + 1];
    for (int c = lane; c < 128; c += 32) {
        float v = (float)Vbf[(size_t)edge * 128 + c];
        atomicAdd(&hAgg[(size_t)d * 128 + c], v * ((c < 64) ? a0 : a1));
    }
}

// ---------- out projection + relu + layernorm ----------
__global__ __launch_bounds__(256) void out_kernel(const float* __restrict__ hAgg, const float* __restrict__ node_h,
                                                  const __bf16* __restrict__ wo_sw, const float* __restrict__ bout,
                                                  const float* __restrict__ gamma, const float* __restrict__ beta,
                                                  float* __restrict__ out, int nDst) {
    __shared__ __bf16 aT[16 * 256];
    __shared__ float  rT[16 * 128];
    __shared__ float  muT[16], rsT[16];
    int tid = threadIdx.x;
    int r0 = blockIdx.x * 16;
    for (int idx = tid; idx < 16 * 256; idx += 256) {
        int row = idx >> 8, col = idx & 255;
        int rr = r0 + row; if (rr >= nDst) rr = nDst - 1;
        float v = (col < 128) ? hAgg[(size_t)rr * 128 + col] : node_h[(size_t)rr * 128 + (col - 128)];
        aT[idx] = (__bf16)v;
    }
    __syncthreads();
    int wave = tid >> 5, lane = tid & 31, g = lane >> 4, rowm = lane & 15;
    v8f c = {};
    for (int kc = 0; kc < 8; ++kc) {
        v16bf a, b;
        int kb = kc * 32;
        #pragma unroll
        for (int e = 0; e < 8; ++e) {
            a[e]     = aT[rowm * 256 + kb + g * 8 + e];
            a[e + 8] = aT[rowm * 256 + kb + 16 + g * 8 + e];
        }
        const __bf16* pb = wo_sw + ((size_t)(kc * 8 + wave) * 32 + lane) * 16;
        #pragma unroll
        for (int e = 0; e < 16; ++e) b[e] = pb[e];
        c = __builtin_amdgcn_wmma_f32_16x16x32_bf16(false, a, false, b, (short)0, c, false, false);
    }
    int ncol = wave * 16 + (lane & 15);
    float bo = bout[ncol];
    #pragma unroll
    for (int r = 0; r < 8; ++r) {
        float v = fmaxf(c[r] + bo, 0.0f);                // relu
        rT[(r + g * 8) * 128 + ncol] = v;
    }
    __syncthreads();
    if (tid < 16) {
        float mu = 0.0f;
        for (int j = 0; j < 128; ++j) mu += rT[tid * 128 + j];
        mu *= (1.0f / 128.0f);
        float var = 0.0f;
        for (int j = 0; j < 128; ++j) { float d = rT[tid * 128 + j] - mu; var += d * d; }
        var *= (1.0f / 128.0f);
        muT[tid] = mu; rsT[tid] = rsqrtf(var + 1e-5f);
    }
    __syncthreads();
    for (int idx = tid; idx < 16 * 128; idx += 256) {
        int row = idx >> 7, col = idx & 127;
        int rr = r0 + row;
        if (rr < nDst) out[(size_t)rr * 128 + col] = (rT[idx] - muT[row]) * rsT[row] * gamma[col] + beta[col];
    }
}

// ---------- launch ----------
extern "C" void kernel_launch(void* const* d_in, const int* in_sizes, int n_in,
                              void* d_out, int out_size, void* d_ws, size_t ws_size,
                              hipStream_t stream) {
    const float* node_h  = (const float*)d_in[0];
    const float* edge_f  = (const float*)d_in[1];
    const float* dt      = (const float*)d_in[2];
    const int*   dst_idx = (const int*)d_in[3];
    const float* time_w  = (const float*)d_in[4];
    const float* time_b  = (const float*)d_in[5];
    const float* Wq   = (const float*)d_in[6];
    const float* bq   = (const float*)d_in[7];
    const float* Wk   = (const float*)d_in[8];
    const float* bk   = (const float*)d_in[9];
    const float* Wv   = (const float*)d_in[10];
    const float* bv   = (const float*)d_in[11];
    const float* Wout = (const float*)d_in[12];
    const float* bout = (const float*)d_in[13];
    const float* gamma = (const float*)d_in[14];
    const float* beta  = (const float*)d_in[15];
    float* out = (float*)d_out;

    const int E    = in_sizes[1] / 128;
    const int nDst = in_sizes[0] / 128 - E;

    char* ws = (char*)d_ws;
    size_t off = 0;
    auto wsAlloc = [&](size_t bytes) -> void* {
        void* p = ws + off;
        off += (bytes + 255) & ~(size_t)255;
        return p;
    };
    __bf16*  wq_sw  = (__bf16*)wsAlloc((size_t)4  * 8 * 512 * sizeof(__bf16));
    __bf16*  wk_sw  = (__bf16*)wsAlloc((size_t)12 * 8 * 512 * sizeof(__bf16));
    __bf16*  wv_sw  = (__bf16*)wsAlloc((size_t)12 * 8 * 512 * sizeof(__bf16));
    __bf16*  wo_sw  = (__bf16*)wsAlloc((size_t)8  * 8 * 512 * sizeof(__bf16));
    float*   qconst = (float*)wsAlloc(128 * sizeof(float));
    float*   Qn     = (float*)wsAlloc((size_t)nDst * 128 * sizeof(float));
    float*   score  = (float*)wsAlloc((size_t)E * 2 * sizeof(float));
    float*   eArr   = (float*)wsAlloc((size_t)E * 2 * sizeof(float));
    unsigned* mKey  = (unsigned*)wsAlloc((size_t)nDst * 2 * sizeof(unsigned));
    float*   sSum   = (float*)wsAlloc((size_t)nDst * 2 * sizeof(float));
    float*   hAgg   = (float*)wsAlloc((size_t)nDst * 128 * sizeof(float));
    __bf16*  Vbf    = (__bf16*)wsAlloc((size_t)E * 128 * sizeof(__bf16));

    // prep: weight swizzle (bf16 fragments) + fused Q constant
    swizzle_kernel<<<32, 256, 0, stream>>>(Wq,   256, 0, wq_sw);
    swizzle_kernel<<<96, 256, 0, stream>>>(Wk,   384, 0, wk_sw);
    swizzle_kernel<<<96, 256, 0, stream>>>(Wv,   384, 0, wv_sw);
    swizzle_kernel<<<64, 256, 0, stream>>>(Wout, 256, 0, wo_sw);
    qconst_kernel<<<1, 128, 0, stream>>>(Wq, bq, time_b, qconst);

    // init accumulators (mKey zero == "empty" in key space)
    zero_kernel<<<(nDst * 2 + 255) / 256, 256, 0, stream>>>((float*)mKey, nDst * 2);
    zero_kernel<<<(nDst * 2 + 255) / 256, 256, 0, stream>>>(sSum, nDst * 2);
    zero_kernel<<<(nDst * 128 + 255) / 256, 256, 0, stream>>>(hAgg, nDst * 128);

    // pipeline
    q_kernel<<<(nDst + 15) / 16, 256, 0, stream>>>(node_h, wq_sw, qconst, Qn, nDst);
    kv_score_kernel<<<(E + 15) / 16, 256, 0, stream>>>(node_h, edge_f, dt, dst_idx, time_w, time_b,
                                                       bk, bv, wk_sw, wv_sw, Qn, score, mKey, Vbf, nDst, E);
    exp_kernel<<<(E * 2 + 255) / 256, 256, 0, stream>>>(score, dst_idx, mKey, eArr, sSum, E);
    agg_kernel<<<(E + 7) / 8, 256, 0, stream>>>(eArr, sSum, dst_idx, Vbf, hAgg, E);
    out_kernel<<<(nDst + 15) / 16, 256, 0, stream>>>(hAgg, node_h, wo_sw, bout, gamma, beta, out, nDst);
}
